// GVPMPNN_57887569215656
// MI455X (gfx1250) — matrix operand order repaired and checked by reading, not compiled, not run on hardware
//
#include <hip/hip_runtime.h>
#include <hip/hip_bf16.h>

// ---------------------------------------------------------------------------
// GVP-MPNN for MI455X (gfx1250, wave32, WMMA 16x16x32 f16).
//   edge kernel : 16 edges per wave, 3 fused GVP layers, all matmuls via WMMA
//   prep kernel : fp32 weights -> f16 transposed, zero-padded panels in d_ws
//   node kernel : mean-agg + LN + dh0/dh1 GVPs + LN (tiny, VALU)
// Transcendentals use hardware v_sqrt_f32 / v_rcp_f32 / v_rsq_f32 (1-ulp),
// keeping the VALU tail short so TRANS/WMMA co-execution is not starved.
// ---------------------------------------------------------------------------

typedef __attribute__((ext_vector_type(16))) _Float16 v16h;
typedef __attribute__((ext_vector_type(2)))  _Float16 h2;
typedef __attribute__((ext_vector_type(8)))  float    v8f;

#define FN 100
#define VN 16
#define FE 32
#define FM 132          // FN+FE
#define KS 180          // FM + 48 (s = concat(feats, sh))
#define LDF 192         // fbuf stride (K padded to 6 k-tiles of 32)
#define LDV 64          // vbuf stride (v_in padded)
#define NSW 144         // scalar weight panel N pad (9 n-tiles)
#define WAVES 4

__device__ __forceinline__ float fsqrt(float x)  { return __builtin_amdgcn_sqrtf(x); }
__device__ __forceinline__ float frcp(float x)   { return __builtin_amdgcn_rcpf(x); }
__device__ __forceinline__ float frsq(float x)   { return __builtin_amdgcn_rsqf(x); }
__device__ __forceinline__ float fsig(float x)   { return frcp(1.f + __expf(-x)); }

__device__ __forceinline__ v8f v8f_zero() {
  v8f r;
#pragma unroll
  for (int i = 0; i < 8; ++i) r[i] = 0.f;
  return r;
}

__device__ __forceinline__ v8f wmma16(v16h a, v16h b, v8f c) {
  // (neg_a, A, neg_b, B, c_mod, C, reuse_a, reuse_b)
  return __builtin_amdgcn_wmma_f32_16x16x32_f16(false, a, false, b, (short)0, c,
                                                false, false);
}

// A fragment: 16xK row-major tile in LDS (stride ld halfs), K-tile kt (32 wide).
// Contiguous K order (same as glbBT): lane m=lane%16, lane-half lh picks K 16*lh..16*lh+15.
__device__ __forceinline__ v16h ldsA(const _Float16* base, int ld, int kt) {
  const int lane = threadIdx.x & 31;
  const int m  = lane & 15;
  const int lh = lane >> 4;
  const _Float16* p = base + m * ld + kt * 32 + 16 * lh;
  v16h r;
#pragma unroll
  for (int g = 0; g < 8; ++g) {
    int rel = 2 * g;   // K = 2g(+16lh), contiguous
    h2 t = *(const h2*)(p + rel);
    r[2 * g]     = t.x;
    r[2 * g + 1] = t.y;
  }
  return r;
}

// B fragment from transposed (N-major) f16 weight panel baseT[n*ldT + k].
// Lane n = lane%16 (+16*nt), K = kt*32 + 16*(lane>>4) + e  -> one contiguous load.
__device__ __forceinline__ v16h glbBT(const _Float16* baseT, int ldT, int kt, int nt) {
  const int lane = threadIdx.x & 31;
  const int nn = nt * 16 + (lane & 15);
  const int k0 = kt * 32 + (lane >> 4) * 16;
  return *(const v16h*)(baseT + nn * ldT + k0);
}

// ---------------------------------------------------------------------------
// prep: convert/transpose/pad weights to f16 panels; Wc = Wh @ Wu fused.
// ---------------------------------------------------------------------------
__global__ void gvp_prep(const float* Wh0, const float* Wu0, const float* W0,
                         const float* Wh1, const float* Wu1, const float* W1,
                         const float* Wh2, const float* Wu2, const float* W2,
                         _Float16* WhPT, _Float16* WcPT, _Float16* WsPT) {
  const int tid = threadIdx.x;
  const float* Whs[3] = {Wh0, Wh1, Wh2};
  const float* Wus[3] = {Wu0, Wu1, Wu2};
  const float* Wss[3] = {W0, W1, W2};
  for (int l = 0; l < 3; ++l) {
    const int vin = (l == 0) ? 17 : 48;
    const float* Wh = Whs[l]; const float* Wu = Wus[l]; const float* W = Wss[l];
    for (int idx = tid; idx < 64 * 64; idx += 256) {
      int n = idx >> 6, k = idx & 63;                 // panel is [n][k]
      float wh = 0.f, wc = 0.f;
      if (k < vin && n < 48) {
        wh = Wh[k * 48 + n];
        float a = 0.f;
        for (int h = 0; h < 48; ++h) a += Wh[k * 48 + h] * Wu[h * 48 + n];
        wc = a;
      }
      WhPT[l * 4096 + idx] = (_Float16)wh;
      WcPT[l * 4096 + idx] = (_Float16)wc;
    }
    for (int idx = tid; idx < NSW * LDF; idx += 256) {
      int n = idx / LDF, k = idx % LDF;
      float w = (k < KS && n < FM) ? W[k * FM + n] : 0.f;
      WsPT[l * (NSW * LDF) + idx] = (_Float16)w;
    }
  }
}

__global__ void gvp_zero(float* p, int n) {
  int i = blockIdx.x * blockDim.x + threadIdx.x;
  if (i < n) p[i] = 0.f;
}

// ---------------------------------------------------------------------------
// edge kernel: 4 waves/block, each wave owns a 16-edge tile and its LDS slice.
// ---------------------------------------------------------------------------
__launch_bounds__(128)
__global__ void gvp_edge(const float* __restrict__ feats_node,
                         const float* __restrict__ vectors_node,
                         const float* __restrict__ feats_edge,
                         const float* __restrict__ vectors_edge,
                         const float* __restrict__ b0, const float* __restrict__ b1,
                         const float* __restrict__ b2,
                         const _Float16* __restrict__ WhPT,
                         const _Float16* __restrict__ WcPT,
                         const _Float16* __restrict__ WsPT,
                         float* __restrict__ accF, float* __restrict__ accV) {
  __shared__ _Float16 shF[WAVES][16 * LDF];      // concat(feats, sh), K-padded
  __shared__ _Float16 shV[WAVES][3 * 16 * LDV];  // per-channel edge vectors

  const int lane = threadIdx.x & 31;
  const int wave = threadIdx.x >> 5;
  const int t  = blockIdx.x * WAVES + wave;      // tile id, 8192 total
  const int b  = t >> 12;
  const int i  = (t >> 4) & 255;
  const int j0 = (t & 15) << 4;

  _Float16* fb = &shF[wave][0];
  _Float16* vb = &shV[wave][0];

  // ---- stage inputs (f16), zero pads --------------------------------------
  for (int idx = lane; idx < 16 * LDF; idx += 32) {
    int m = idx / LDF, col = idx % LDF;
    float val = 0.f;
    if (col < FN)      val = feats_node[(b * 256 + j0 + m) * FN + col];
    else if (col < FM) val = feats_edge[((b * 256 + i) * 256 + j0 + m) * FE + (col - FN)];
    fb[m * LDF + col] = (_Float16)val;
  }
  for (int idx = lane; idx < 3 * 16 * LDV; idx += 32) {
    int c = idx / (16 * LDV);
    int r = idx % (16 * LDV);
    int m = r / LDV, v = r % LDV;
    float val = 0.f;
    if (v < VN)       val = vectors_node[(b * 256 + j0 + m) * (VN * 3) + v * 3 + c];
    else if (v == VN) val = vectors_edge[((b * 256 + i) * 256 + j0 + m) * 3 + c];
    vb[idx] = (_Float16)val;
  }

  // ---- 3 fused GVP layers -------------------------------------------------
#pragma unroll
  for (int l = 0; l < 3; ++l) {
    const int kt = (l == 0) ? 1 : 2;              // K tiles for vector path
    const _Float16* WhL = WhPT + l * 4096;
    const _Float16* WcL = WcPT + l * 4096;
    const _Float16* WsL = WsPT + l * (NSW * LDF);
    const float* bL = (l == 0) ? b0 : ((l == 1) ? b1 : b2);
    __builtin_prefetch(WsL, 0, 3);                // global_prefetch_b8 into L2

    // vector path: Vh (for sh) and Vu = X @ (Wh@Wu), per xyz channel
    v16h Av[3][2];
#pragma unroll
    for (int c = 0; c < 3; ++c)
#pragma unroll
      for (int k = 0; k < 2; ++k)
        if (k < kt) Av[c][k] = ldsA(vb + c * 16 * LDV, LDV, k);

#pragma unroll
    for (int nt = 0; nt < 3; ++nt) {
      v8f s2 = v8f_zero();
      v8f vu[3];
#pragma unroll
      for (int c = 0; c < 3; ++c) {
        v8f acc = v8f_zero();
#pragma unroll
        for (int k = 0; k < 2; ++k)
          if (k < kt) acc = wmma16(Av[c][k], glbBT(WhL, 64, k, nt), acc);
        s2 += acc * acc;
        v8f acc2 = v8f_zero();
#pragma unroll
        for (int k = 0; k < 2; ++k)
          if (k < kt) acc2 = wmma16(Av[c][k], glbBT(WcL, 64, k, nt), acc2);
        vu[c] = acc2;
      }
      const int nn = lane & 15, lh = lane >> 4;
#pragma unroll
      for (int e = 0; e < 8; ++e) {
        int m = e + 8 * lh;
        int u = nt * 16 + nn;
        float g2 = vu[0][e] * vu[0][e] + vu[1][e] * vu[1][e] + vu[2][e] * vu[2][e];
        float gate = fsig(fsqrt(g2));
        vb[0 * 16 * LDV + m * LDV + u] = (_Float16)(gate * vu[0][e]);
        vb[1 * 16 * LDV + m * LDV + u] = (_Float16)(gate * vu[1][e]);
        vb[2 * 16 * LDV + m * LDV + u] = (_Float16)(gate * vu[2][e]);
        fb[m * LDF + FM + u] = (_Float16)fsqrt(s2[e]);   // sh into s columns
      }
    }

    // scalar path: feats_out = sigmoid(s @ W + b), s = fb rows (K=180 pad 192)
    v16h As[6];
#pragma unroll
    for (int k = 0; k < 6; ++k) As[k] = ldsA(fb, LDF, k);
#pragma unroll
    for (int nt = 0; nt < 9; ++nt) {
      v8f acc = v8f_zero();
#pragma unroll
      for (int k = 0; k < 6; ++k)
        acc = wmma16(As[k], glbBT(WsL, LDF, k, nt), acc);
      const int nn = lane & 15, lh = lane >> 4;
      const int col = nt * 16 + nn;
      if (col < FM) {
        float bv = bL[col];
#pragma unroll
        for (int e = 0; e < 8; ++e) {
          int m = e + 8 * lh;
          fb[m * LDF + col] = (_Float16)fsig(acc[e] + bv);
        }
      }
    }
  }

  // ---- mean-aggregate this tile's 16 j's into (b,i) accumulators ----------
  const int node = b * 256 + i;
  for (int col = lane; col < FN; col += 32) {
    float s = 0.f;
#pragma unroll
    for (int m = 0; m < 16; ++m) s += (float)fb[m * LDF + col];
    atomicAdd(&accF[node * FN + col], s);
  }
  for (int idx = lane; idx < VN * 3; idx += 32) {
    int v = idx / 3, c = idx % 3;
    float s = 0.f;
#pragma unroll
    for (int m = 0; m < 16; ++m) s += (float)vb[c * 16 * LDV + m * LDV + v];
    atomicAdd(&accV[(node * VN + v) * 3 + c], s);
  }
}

// ---------------------------------------------------------------------------
// node kernel: one 128-thread block per node (512 nodes)
// ---------------------------------------------------------------------------
#define NT 128
__device__ __forceinline__ float blockSum(float v, float* red) {
  int tid = threadIdx.x;
  red[tid] = v; __syncthreads();
  for (int s = NT / 2; s > 0; s >>= 1) {
    if (tid < s) red[tid] += red[tid + s];
    __syncthreads();
  }
  float r = red[0]; __syncthreads();
  return r;
}

__global__ void gvp_node(const float* __restrict__ feats_node,
                         const float* __restrict__ vectors_node,
                         const float* __restrict__ accF, const float* __restrict__ accV,
                         const float* Wh0, const float* Wu0, const float* W0, const float* b0,
                         const float* Wh1, const float* Wu1, const float* W1, const float* b1,
                         const float* g0, const float* be0,
                         const float* g1, const float* be1,
                         float* __restrict__ outF, float* __restrict__ outV) {
  __shared__ float red[NT];
  __shared__ float sFeat[FN];
  __shared__ float sVec[3][16];
  __shared__ float sS[432];
  __shared__ float sVh[3][32];
  __shared__ float sVu[3][32];
  __shared__ float sF2[400];
  __shared__ float sV2[3][32];
  __shared__ float sFr[FN];
  __shared__ float sVr[3][16];

  const int tid  = threadIdx.x;
  const int node = blockIdx.x;

  for (int c = tid; c < FN; c += NT)
    sS[c] = accF[node * FN + c] * (1.f / 256.f) + feats_node[node * FN + c];
  for (int idx = tid; idx < 48; idx += NT) {
    int v = idx / 3, c = idx % 3;
    sVec[c][v] = accV[(node * 16 + v) * 3 + c] * (1.f / 256.f)
               + vectors_node[(node * 16 + v) * 3 + c];
  }
  __syncthreads();

  // LN0
  float p = 0.f;
  for (int c = tid; c < FN; c += NT) p += sS[c];
  float mu = blockSum(p, red) * (1.f / FN);
  p = 0.f;
  for (int c = tid; c < FN; c += NT) { float d = sS[c] - mu; p += d * d; }
  float inv = frsq(blockSum(p, red) * (1.f / FN) + 1e-5f);
  for (int c = tid; c < FN; c += NT) sFeat[c] = (sS[c] - mu) * inv * g0[c] + be0[c];
  p = 0.f;
  for (int idx = tid; idx < 48; idx += NT) { int v = idx / 3, c = idx % 3; float x = sVec[c][v]; p += x * x; }
  float rvn = frcp(fsqrt(blockSum(p, red)) + 1e-8f);
  for (int idx = tid; idx < 48; idx += NT) { int v = idx / 3, c = idx % 3; sVec[c][v] *= rvn; }
  __syncthreads();

  // dh0
  for (int idx = tid; idx < 96; idx += NT) {
    int c = idx >> 5, h = idx & 31;
    float a = 0.f;
    for (int v = 0; v < 16; ++v) a += sVec[c][v] * Wh0[v * 32 + h];
    sVh[c][h] = a;
  }
  __syncthreads();
  for (int idx = tid; idx < 96; idx += NT) {
    int c = idx >> 5, u = idx & 31;
    float a = 0.f;
    for (int h = 0; h < 32; ++h) a += sVh[c][h] * Wu0[h * 32 + u];
    sVu[c][u] = a;
  }
  for (int idx = tid; idx < 132; idx += NT) {
    int h = idx - FN;
    sS[idx] = (idx < FN) ? sFeat[idx]
            : fsqrt(sVh[0][h]*sVh[0][h] + sVh[1][h]*sVh[1][h] + sVh[2][h]*sVh[2][h]);
  }
  __syncthreads();
  for (int o = tid; o < 400; o += NT) {
    float a = b0[o];
    for (int k = 0; k < 132; ++k) a += sS[k] * W0[k * 400 + o];
    sF2[o] = fsig(a);
  }
  for (int idx = tid; idx < 96; idx += NT) {
    int c = idx >> 5, u = idx & 31;
    float g2 = sVu[0][u]*sVu[0][u] + sVu[1][u]*sVu[1][u] + sVu[2][u]*sVu[2][u];
    float gate = fsig(fsqrt(g2));
    sV2[c][u] = gate * sVu[c][u];
  }
  __syncthreads();

  // dh1
  for (int idx = tid; idx < 96; idx += NT) {
    int c = idx >> 5, h = idx & 31;
    float a = 0.f;
    for (int v = 0; v < 32; ++v) a += sV2[c][v] * Wh1[v * 32 + h];
    sVh[c][h] = a;
  }
  __syncthreads();
  for (int idx = tid; idx < 432; idx += NT) {
    int h = idx - 400;
    sS[idx] = (idx < 400) ? sF2[idx]
            : fsqrt(sVh[0][h]*sVh[0][h] + sVh[1][h]*sVh[1][h] + sVh[2][h]*sVh[2][h]);
  }
  for (int idx = tid; idx < 48; idx += NT) {
    int c = idx >> 4, u = idx & 15;
    float a = 0.f;
    for (int h = 0; h < 32; ++h) a += sVh[c][h] * Wu1[h * 16 + u];
    sVu[c][u] = a;
  }
  __syncthreads();
  for (int o = tid; o < FN; o += NT) {
    float a = b1[o];
    for (int k = 0; k < 432; ++k) a += sS[k] * W1[k * FN + o];
    sFr[o] = fsig(a) + sFeat[o];
  }
  for (int idx = tid; idx < 48; idx += NT) {
    int c = idx >> 4, u = idx & 15;
    float g2 = sVu[0][u]*sVu[0][u] + sVu[1][u]*sVu[1][u] + sVu[2][u]*sVu[2][u];
    float gate = fsig(fsqrt(g2));
    sVr[c][u] = gate * sVu[c][u] + sVec[c][u];
  }
  __syncthreads();

  // LN1 + outputs
  p = 0.f;
  for (int o = tid; o < FN; o += NT) p += sFr[o];
  mu = blockSum(p, red) * (1.f / FN);
  p = 0.f;
  for (int o = tid; o < FN; o += NT) { float d = sFr[o] - mu; p += d * d; }
  inv = frsq(blockSum(p, red) * (1.f / FN) + 1e-5f);
  for (int o = tid; o < FN; o += NT)
    outF[node * FN + o] = (sFr[o] - mu) * inv * g1[o] + be1[o];
  p = 0.f;
  for (int idx = tid; idx < 48; idx += NT) { int v = idx / 3, c = idx % 3; float x = sVr[c][v]; p += x * x; }
  rvn = frcp(fsqrt(blockSum(p, red)) + 1e-8f);
  for (int idx = tid; idx < 48; idx += NT) {
    int v = idx / 3, c = idx % 3;
    outV[(node * 16 + v) * 3 + c] = sVr[c][v] * rvn;
  }
}

// ---------------------------------------------------------------------------
// host
// ---------------------------------------------------------------------------
extern "C" void kernel_launch(void* const* d_in, const int* in_sizes, int n_in,
                              void* d_out, int out_size, void* d_ws, size_t ws_size,
                              hipStream_t stream) {
  // locate tensors by flat element counts (works for insertion or sorted order)
  int iFN = -1, iVN = -1, iFE = -1, iVE = -1;
  int iW[5] = {-1, -1, -1, -1, -1};   // ev0,ev1,ev2,dh0,dh1 "W"
  int nev = 0;
  for (int i = 0; i < n_in; ++i) {
    int s = in_sizes[i];
    if      (s == 51200)   iFN = i;
    else if (s == 24576)   iVN = i;
    else if (s == 4194304) iFE = i;
    else if (s == 393216)  iVE = i;
    else if (s == 23760) { if (nev < 3) iW[nev] = i; ++nev; }
    else if (s == 52800)   iW[3] = i;
    else if (s == 43200)   iW[4] = i;
  }
  const int bsz[5] = {132, 132, 132, 400, 100};
  int iWh[5], iWu[5], iB[5];
  bool sortedLayout = false;
  for (int g = 0; g < 5; ++g) {
    int w = iW[g];
    if (w + 1 < n_in && in_sizes[w + 1] == bsz[g]) {      // Wh,Wu,W,b
      iWh[g] = w - 2; iWu[g] = w - 1; iB[g] = w + 1;
    } else {                                              // W,Wh,Wu,b (sorted)
      iWh[g] = w + 1; iWu[g] = w + 2; iB[g] = w + 3;
      sortedLayout = true;
    }
  }
  int lnIdx[4]; int nc = 0;
  for (int i = 0; i < n_in && nc < 4; ++i)
    if (in_sizes[i] == 100 && i != iB[4]) lnIdx[nc++] = i;
  int iG0, iBe0, iG1, iBe1;
  if (sortedLayout) { iBe0 = lnIdx[0]; iG0 = lnIdx[1]; iBe1 = lnIdx[2]; iG1 = lnIdx[3]; }
  else              { iG0 = lnIdx[0]; iBe0 = lnIdx[1]; iG1 = lnIdx[2]; iBe1 = lnIdx[3]; }

  const float* fN = (const float*)d_in[iFN];
  const float* vN = (const float*)d_in[iVN];
  const float* fE = (const float*)d_in[iFE];
  const float* vE = (const float*)d_in[iVE];

  // workspace layout
  float* accF = (float*)d_ws;                 // 2*256*100
  float* accV = accF + 51200;                 // 2*256*16*3
  _Float16* WhPT = (_Float16*)(accV + 24576); // 3 * 64*64
  _Float16* WcPT = WhPT + 3 * 4096;           // 3 * 64*64
  _Float16* WsPT = WcPT + 3 * 4096;           // 3 * 144*192

  gvp_prep<<<1, 256, 0, stream>>>(
      (const float*)d_in[iWh[0]], (const float*)d_in[iWu[0]], (const float*)d_in[iW[0]],
      (const float*)d_in[iWh[1]], (const float*)d_in[iWu[1]], (const float*)d_in[iW[1]],
      (const float*)d_in[iWh[2]], (const float*)d_in[iWu[2]], (const float*)d_in[iW[2]],
      WhPT, WcPT, WsPT);

  gvp_zero<<<(75776 + 255) / 256, 256, 0, stream>>>(accF, 75776);

  gvp_edge<<<8192 / WAVES, 32 * WAVES, 0, stream>>>(
      fN, vN, fE, vE,
      (const float*)d_in[iB[0]], (const float*)d_in[iB[1]], (const float*)d_in[iB[2]],
      WhPT, WcPT, WsPT, accF, accV);

  float* outF = (float*)d_out;
  float* outV = outF + 51200;
  gvp_node<<<512, NT, 0, stream>>>(
      fN, vN, accF, accV,
      (const float*)d_in[iWh[3]], (const float*)d_in[iWu[3]],
      (const float*)d_in[iW[3]],  (const float*)d_in[iB[3]],
      (const float*)d_in[iWh[4]], (const float*)d_in[iWu[4]],
      (const float*)d_in[iW[4]],  (const float*)d_in[iB[4]],
      (const float*)d_in[iG0], (const float*)d_in[iBe0],
      (const float*)d_in[iG1], (const float*)d_in[iBe1],
      outF, outV);
}
